// ScaledDotProductAttention_8100308320896
// MI455X (gfx1250) — matrix operand order repaired and checked by reading, not compiled, not run on hardware
//
#include <hip/hip_runtime.h>

typedef __attribute__((ext_vector_type(16))) __bf16 v16bf;
typedef __attribute__((ext_vector_type(8)))  __bf16 v8bf;
typedef __attribute__((ext_vector_type(8)))  float  v8f;
typedef __attribute__((ext_vector_type(4)))  float  v4f;

#define B_    16
#define S_    2048
#define D_    128
#define QTILE 64   // 4 waves * 16 rows

// v_wmma_f32_16x16x32_bf16: D = A(16x32) * B(32x16) + C(16x16 f32), wave32.
// A layout (ISA 7.12.2): lane l -> row m = l%16; elem j<8 -> K = 8*(l/16)+j,
//                        elem j>=8 -> K = 16 + 8*(l/16) + (j-8)
// B as loaded here:      lane l -> col n = l%16; elem j -> K = 16*(l/16)+j (A here too)
// C/D layout:            lane l -> col n = l%16; VGPR r -> row m = r + 8*(l/16)

__device__ __forceinline__ unsigned int pk_bf16x2(float lo, float hi) {
    unsigned short a = __builtin_bit_cast(unsigned short, (__bf16)lo);
    unsigned short b = __builtin_bit_cast(unsigned short, (__bf16)hi);
    return ((unsigned int)b << 16) | (unsigned int)a;   // -> v_cvt_pk_bf16_f32
}

__global__ __launch_bounds__(128) void sdpa_fwd_kernel(
    const float* __restrict__ q, const float* __restrict__ k,
    const float* __restrict__ v, float* __restrict__ ctx,
    float* __restrict__ attn)
{
    __shared__ __bf16 ldsK[32 * D_];      // K tile, row-major KS[n][d], 8 KB
    __shared__ __bf16 ldsVT[D_ * 32];     // transposed V tile VT[d][n],  8 KB
    __shared__ __bf16 ldsP[4][16 * 32];   // per-wave P staging,          4 KB

    const int tid  = threadIdx.x;
    const int wave = tid >> 5;
    const int lane = tid & 31;
    const int lg   = lane >> 4;           // lane half-group
    const int ln   = lane & 15;

    const int b  = blockIdx.x >> 5;             // 32 q-tiles per batch
    const int q0 = (blockIdx.x & 31) * QTILE;
    const int qm = q0 + wave * 16;              // first row of this wave's strip

    const float* qrow = q + ((size_t)b * S_ + (qm + ln)) * (size_t)D_;
    const float* kb   = k + (size_t)b * S_ * D_;
    const float* vb   = v + (size_t)b * S_ * D_;

    // log2-domain softmax: scores scaled by (1/sqrt(128)) * log2(e); softmax is
    // invariant under the base change, exp2 == raw v_exp_f32 (no extra v_mul).
    const float scale2 = 0.08838834764831845f * 1.4426950408889634f;

    // ---- resident Q strip: 4 bf16 A-fragments (16x32 each) -----------------
    v16bf aq[4];
#pragma unroll
    for (int i = 0; i < 4; ++i) {
        const int d0 = 32 * i + 16 * lg;
#pragma unroll
        for (int j = 0; j < 8; ++j) aq[i][j]     = (__bf16)qrow[d0 + j];
#pragma unroll
        for (int j = 0; j < 8; ++j) aq[i][8 + j] = (__bf16)qrow[d0 + 8 + j];
    }

    // cooperative stage: K[n0..n0+32) x [0..128) f32 -> bf16 LDS row-major
    auto stage_k = [&](int n0) {
#pragma unroll
        for (int i = 0; i < 8; ++i) {
            const int n = 4 * i + (tid >> 5);
            const int d = (tid & 31) * 4;
            v4f f = *(const v4f*)(kb + (size_t)(n0 + n) * D_ + d);
            *(unsigned int*)&ldsK[n * D_ + d]     = pk_bf16x2(f[0], f[1]);
            *(unsigned int*)&ldsK[n * D_ + d + 2] = pk_bf16x2(f[2], f[3]);
        }
    };

    // cooperative stage: V tile transposed, VT[d][n] (row pairs packed b32)
    auto stage_v = [&](int n0) {
#pragma unroll
        for (int i = 0; i < 4; ++i) {
            const int n = 8 * i + (tid >> 5) * 2;   // even row index
            const int d = (tid & 31) * 4;
            v4f fa = *(const v4f*)(vb + (size_t)(n0 + n)     * D_ + d);
            v4f fb = *(const v4f*)(vb + (size_t)(n0 + n + 1) * D_ + d);
#pragma unroll
            for (int c = 0; c < 4; ++c)
                *(unsigned int*)&ldsVT[(d + c) * 32 + n] = pk_bf16x2(fa[c], fb[c]);
        }
    };

    // scores tile from staged K: S2[16x16] = scale2 * Q_strip x KS[half]^T
    auto score_tile = [&](int half) -> v8f {
        v8f c = {};
        const __bf16* kp = &ldsK[(16 * half + ln) * D_ + 16 * lg];
#pragma unroll
        for (int i = 0; i < 4; ++i) {
            v16bf bk = *(const v16bf*)(kp + 32 * i);
            c = __builtin_amdgcn_wmma_f32_16x16x32_bf16(false, aq[i], false, bk,
                                                        (short)0, c, false, false);
        }
#pragma unroll
        for (int r = 0; r < 8; ++r) c[r] *= scale2;
        return c;
    };

    // ---- pass 1: fused lane-local online softmax (no per-tile shuffles) ----
    float rm[8], rcl[8];
    {
        v8f mloc, vsum;
#pragma unroll
        for (int r = 0; r < 8; ++r) { mloc[r] = -3.0e38f; vsum[r] = 0.0f; }

        for (int n0 = 0; n0 < S_; n0 += 32) {
            __syncthreads();
            stage_k(n0);
            __syncthreads();
            if (n0 + 32 < S_)
                __builtin_prefetch((const char*)(kb + (size_t)(n0 + 32) * D_) + tid * 128, 0, 1);
#pragma unroll
            for (int half = 0; half < 2; ++half) {
                v8f s = score_tile(half);
#pragma unroll
                for (int r = 0; r < 8; ++r) {
                    float mnew = fmaxf(mloc[r], s[r]);
                    vsum[r] = vsum[r] * __builtin_amdgcn_exp2f(mloc[r] - mnew)
                            + __builtin_amdgcn_exp2f(s[r] - mnew);
                    mloc[r] = mnew;
                }
            }
        }
        // one-time cross-lane combine (exact): gmax, then rescaled partial sums
#pragma unroll
        for (int r = 0; r < 8; ++r) {
            float gm = mloc[r];
#pragma unroll
            for (int msk = 1; msk < 16; msk <<= 1)
                gm = fmaxf(gm, __shfl_xor(gm, msk, 32));
            float part = vsum[r] * __builtin_amdgcn_exp2f(mloc[r] - gm);
#pragma unroll
            for (int msk = 1; msk < 16; msk <<= 1)
                part += __shfl_xor(part, msk, 32);
            rm[r]  = gm;            // row max (log2 domain), uniform in group
            rcl[r] = 1.0f / part;   // 1 / row sum
        }
    }

    // ---- pass 2: emit normalized attention + accumulate context ------------
    v8f acc[8];
#pragma unroll
    for (int t = 0; t < 8; ++t)
#pragma unroll
        for (int r = 0; r < 8; ++r) acc[t][r] = 0.0f;

    float*  attnrow = attn + ((size_t)b * S_ + qm) * (size_t)S_;
    __bf16* pstage  = &ldsP[wave][0];

    for (int n0 = 0; n0 < S_; n0 += 32) {
        __syncthreads();
        stage_k(n0);
        stage_v(n0);
        __syncthreads();
        if (n0 + 32 < S_) {
            __builtin_prefetch((const char*)(kb + (size_t)(n0 + 32) * D_) + tid * 128, 0, 1);
            __builtin_prefetch((const char*)(vb + (size_t)(n0 + 32) * D_) + tid * 128, 0, 1);
        }

#pragma unroll
        for (int half = 0; half < 2; ++half) {
            const int nh = n0 + 16 * half;
            v8f s = score_tile(half);
#pragma unroll
            for (int r = 0; r < 8; ++r) {
                float p = __builtin_amdgcn_exp2f(s[r] - rm[r]) * rcl[r];
                attnrow[(size_t)(r + 8 * lg) * S_ + nh + ln] = p;       // attention out
                pstage[(r + 8 * lg) * 32 + 16 * half + ln] = (__bf16)p; // stage for PV
            }
        }

        // regather P as a 16x32 A-fragment (wave-private LDS, DS in-order)
        v8bf plo = *(const v8bf*)(pstage + ln * 32 + 16 * lg);
        v8bf phi = *(const v8bf*)(pstage + ln * 32 + 16 * lg + 8);
        v16bf pa;
#pragma unroll
        for (int j = 0; j < 8; ++j) { pa[j] = plo[j]; pa[8 + j] = phi[j]; }

        // context += P(16x32) x V(32x16 per d-tile), 8 d-tiles
#pragma unroll
        for (int t = 0; t < 8; ++t) {
            v16bf bv = *(const v16bf*)&ldsVT[(t * 16 + ln) * 32 + 16 * lg];
            acc[t] = __builtin_amdgcn_wmma_f32_16x16x32_bf16(false, pa, false, bv,
                                                             (short)0, acc[t], false, false);
        }
    }

    // ---- write context strip (f32) -----------------------------------------
    float* cb = ctx + ((size_t)b * S_ + qm) * (size_t)D_;
#pragma unroll
    for (int t = 0; t < 8; ++t)
#pragma unroll
        for (int r = 0; r < 8; ++r)
            cb[(size_t)(r + 8 * lg) * D_ + t * 16 + ln] = acc[t][r];
}

extern "C" void kernel_launch(void* const* d_in, const int* in_sizes, int n_in,
                              void* d_out, int out_size, void* d_ws, size_t ws_size,
                              hipStream_t stream) {
    const float* q = (const float*)d_in[0];
    const float* k = (const float*)d_in[1];
    const float* v = (const float*)d_in[2];
    float* ctx  = (float*)d_out;
    float* attn = ctx + (size_t)B_ * S_ * D_;   // tuple: (context, attention) flat

    dim3 grid(B_ * (S_ / QTILE));   // 512 workgroups
    dim3 block(128);                // 4 waves
    sdpa_fwd_kernel<<<grid, block, 0, stream>>>(q, k, v, ctx, attn);
}
